// acModel_2731599200730
// MI455X (gfx1250) — matrix operand order repaired and checked by reading, not compiled, not run on hardware
//
#include <hip/hip_runtime.h>

// ---------------------------------------------------------------------------
// GRU rollout (B=64, T=512, H=1024, D=63) as one persistent kernel.
// f16 weights/activations, f32 accumulation via v_wmma_f32_16x16x32_f16.
// 16 workgroups x 256 threads (8 waves, wave32). Grid-wide sense barrier
// between the 4 dependent GEMM phases of each time step.
// ---------------------------------------------------------------------------

typedef __attribute__((ext_vector_type(16))) _Float16 v16h;
typedef __attribute__((ext_vector_type(8)))  _Float16 v8h;
typedef __attribute__((ext_vector_type(8)))  float    v8f;

#define NWG 16
#define THREADS 256
#define H 1024
#define B 64
#define T 512
#define D 63
#define DPAD 64

// ---- WMMA fragment loaders -------------------------------------------------
// A (16x32 f16, row-major source, leading dim ld):
//   lane L: row = row0 + (L&15); holds K = kb..kb+7 and kb+16..kb+23,
//   kb = k0 + (L>>4)*8  -> two contiguous 16B loads.
__device__ inline v16h load_a_frag(const _Float16* __restrict__ base, int ld,
                                   int row0, int k0, int lane) {
  int row = row0 + (lane & 15);
  int kb  = k0 + ((lane >> 4) << 3);
  const _Float16* p = base + row * ld + kb;
  union { v16h v; v8h h[2]; } u;
  u.h[0] = *(const v8h*)(p);
  u.h[1] = *(const v8h*)(p + 16);
  return u.v;
}

// B (32x16 f16 = W^T tile). lane L: column n = n0 + (L&15) == W row;
// holds K = kb..kb+7 and kb+16..kb+23, kb = k0 + (L>>4)*8 -> two 16B loads.
__device__ inline v16h load_b_frag(const _Float16* __restrict__ w, int ld,
                                   int n0, int k0, int lane) {
  int r  = n0 + (lane & 15);
  int kb = k0 + ((lane >> 4) << 3);
  const _Float16* p = w + r * ld + kb;
  __builtin_prefetch((const void*)(p + 256), 0, 0); // stream next weight chunk
  union { v16h v; v8h h[2]; } u;
  u.h[0] = *(const v8h*)(p);
  u.h[1] = *(const v8h*)(p + 16);
  return u.v;
}

__device__ inline v8f wmma_f16(v16h a, v16h b, v8f c) {
  return __builtin_amdgcn_wmma_f32_16x16x32_f16(false, a, false, b,
                                                (short)0, c, false, false);
}

// ---- grid-wide barrier (bar[0]=counter, bar[1]=generation) ----------------
__device__ inline void grid_sync(unsigned* bar) {
  __syncthreads();
  if (threadIdx.x == 0) {
    volatile unsigned* vgen = bar + 1;
    unsigned g = *vgen;
    __threadfence();
    if (atomicAdd(bar, 1u) == (unsigned)(NWG - 1)) {
      atomicExch(bar, 0u);
      __threadfence();
      atomicAdd((unsigned*)(bar + 1), 1u);
    } else {
      while (*vgen == g) { __builtin_amdgcn_s_sleep(1); }
      __threadfence();
    }
  }
  __syncthreads();
}

// ---- K-reduction over a 32(batch)x16(col) strip ---------------------------
__device__ inline void gemm_strip(const _Float16* __restrict__ A,
                                  const _Float16* __restrict__ W,
                                  int K, int mb, int n0, int lane, v8f* acc) {
  for (int k = 0; k < K; k += 32) {
    v16h a0 = load_a_frag(A, K, mb,      k, lane);
    v16h a1 = load_a_frag(A, K, mb + 16, k, lane);
    v16h b  = load_b_frag(W, K, n0,      k, lane);
    acc[0] = wmma_f16(a0, b, acc[0]);
    acc[1] = wmma_f16(a1, b, acc[1]);
  }
}

// ---- prep: f32 -> f16 weights (pad K 63->64 for W_ih, rows 63->64 for W3) --
__global__ void prep_weights(const float* __restrict__ Wih,
                             const float* __restrict__ Whh,
                             const float* __restrict__ W1f,
                             const float* __restrict__ W2f,
                             const float* __restrict__ W3f,
                             _Float16* wih16, _Float16* whh16,
                             _Float16* w1h, _Float16* w2h, _Float16* w3h) {
  int stride = gridDim.x * blockDim.x;
  int tid = blockIdx.x * blockDim.x + threadIdx.x;
  for (int i = tid; i < 3 * H * DPAD; i += stride) {
    int r = i >> 6, c = i & 63;
    wih16[i] = (_Float16)((c < D) ? Wih[r * D + c] : 0.f);
  }
  for (int i = tid; i < 3 * H * H; i += stride) whh16[i] = (_Float16)Whh[i];
  for (int i = tid; i < H * H; i += stride) {
    w1h[i] = (_Float16)W1f[i];
    w2h[i] = (_Float16)W2f[i];
  }
  for (int i = tid; i < DPAD * H; i += stride) {
    int r = i >> 10;
    w3h[i] = (_Float16)((r < D) ? W3f[i] : 0.f);
  }
}

// ---- init: h=0, inp = cond(0) ? x[:,0,:] : 0, barrier=0 --------------------
__global__ void init_state(const float* __restrict__ x,
                           const int* gt_p, const int* cl_p,
                           float* h32, _Float16* h16a, _Float16* h16b,
                           float* inp32, _Float16* inp16, unsigned* bar) {
  int stride = gridDim.x * blockDim.x;
  int tid = blockIdx.x * blockDim.x + threadIdx.x;
  int gt = gt_p[0], per = gt + cl_p[0];
  int c0 = ((0 % per) < gt) ? 1 : 0;
  for (int i = tid; i < B * H; i += stride) {
    h32[i] = 0.f; h16a[i] = (_Float16)0.f; h16b[i] = (_Float16)0.f;
  }
  for (int i = tid; i < B * DPAD; i += stride) {
    int r = i >> 6, c = i & 63;
    float v = (c < D && c0) ? x[(r * T + 0) * D + c] : 0.f;
    inp32[i] = v; inp16[i] = (_Float16)v;
  }
  if (tid == 0) { bar[0] = 0u; bar[1] = 0u; }
}

// ---- persistent rollout ----------------------------------------------------
__global__ void __launch_bounds__(THREADS, 1)
gru_rollout(const float* __restrict__ x,
            const _Float16* __restrict__ wih, const _Float16* __restrict__ whh,
            const float* __restrict__ bih, const float* __restrict__ bhh,
            const _Float16* __restrict__ w1, const float* __restrict__ b1,
            const _Float16* __restrict__ w2, const float* __restrict__ b2,
            const _Float16* __restrict__ w3, const float* __restrict__ b3,
            float* __restrict__ out,
            float* h32, _Float16* h16a, _Float16* h16b,
            _Float16* t1h, _Float16* t2h,
            float* inp32, _Float16* inp16,
            unsigned* bar, const int* gt_p, const int* cl_p) {
  const int lane = threadIdx.x & 31;
  const int wave = blockIdx.x * (THREADS >> 5) + (threadIdx.x >> 5); // 0..127
  const int gt = gt_p[0], per = gt + cl_p[0];

  const int j    = wave >> 1;        // column strip 0..63
  const int mb   = (wave & 1) << 5;  // batch-half row base: 0 or 32
  const int jc   = j << 4;           // column base in H
  const int ncol = lane & 15;

  for (int t = 0; t < T; ++t) {
    const _Float16* hin  = (t & 1) ? h16b : h16a; // h_{t-1}
    _Float16*       hout = (t & 1) ? h16a : h16b; // h_t (double-buffered)

    // ===== Phase 1: gate GEMMs + GRU update (r,z,ni,nh in registers) ======
    {
      float br_ = bih[jc + ncol]            + bhh[jc + ncol];
      float bz_ = bih[H + jc + ncol]        + bhh[H + jc + ncol];
      float bni = bih[2 * H + jc + ncol];
      float bnh = bhh[2 * H + jc + ncol];
      v8f ar[2], az[2], ani[2], anh[2];
      for (int m = 0; m < 2; ++m)
        for (int i = 0; i < 8; ++i) {
          ar[m][i] = br_; az[m][i] = bz_; ani[m][i] = bni; anh[m][i] = bnh;
        }
      // hidden part: K = 1024
      for (int k = 0; k < H; k += 32) {
        v16h a0 = load_a_frag(hin, H, mb,      k, lane);
        v16h a1 = load_a_frag(hin, H, mb + 16, k, lane);
        v16h Br = load_b_frag(whh, H, jc,         k, lane);
        v16h Bz = load_b_frag(whh, H, H + jc,     k, lane);
        v16h Bn = load_b_frag(whh, H, 2 * H + jc, k, lane);
        ar[0]  = wmma_f16(a0, Br, ar[0]);  ar[1]  = wmma_f16(a1, Br, ar[1]);
        az[0]  = wmma_f16(a0, Bz, az[0]);  az[1]  = wmma_f16(a1, Bz, az[1]);
        anh[0] = wmma_f16(a0, Bn, anh[0]); anh[1] = wmma_f16(a1, Bn, anh[1]);
      }
      // input part: K = 64 (63 padded)
      for (int k = 0; k < DPAD; k += 32) {
        v16h a0 = load_a_frag(inp16, DPAD, mb,      k, lane);
        v16h a1 = load_a_frag(inp16, DPAD, mb + 16, k, lane);
        v16h Br = load_b_frag(wih, DPAD, jc,         k, lane);
        v16h Bz = load_b_frag(wih, DPAD, H + jc,     k, lane);
        v16h Bn = load_b_frag(wih, DPAD, 2 * H + jc, k, lane);
        ar[0]  = wmma_f16(a0, Br, ar[0]);  ar[1]  = wmma_f16(a1, Br, ar[1]);
        az[0]  = wmma_f16(a0, Bz, az[0]);  az[1]  = wmma_f16(a1, Bz, az[1]);
        ani[0] = wmma_f16(a0, Bn, ani[0]); ani[1] = wmma_f16(a1, Bn, ani[1]);
      }
      for (int m = 0; m < 2; ++m)
        for (int i = 0; i < 8; ++i) {
          int row = mb + (m << 4) + i + ((lane >> 4) << 3);
          int col = jc + ncol;
          float r  = 1.f / (1.f + __expf(-ar[m][i]));
          float z  = 1.f / (1.f + __expf(-az[m][i]));
          float nn = tanhf(ani[m][i] + r * anh[m][i]);
          float hp = h32[row * H + col];
          float hv = (1.f - z) * nn + z * hp;
          h32[row * H + col]  = hv;
          hout[row * H + col] = (_Float16)hv;
        }
    }
    grid_sync(bar);

    // ===== Phase 2: t1 = h_t @ W1^T + b1 ==================================
    {
      v8f acc[2];
      float bb = b1[jc + ncol];
      for (int m = 0; m < 2; ++m) for (int i = 0; i < 8; ++i) acc[m][i] = bb;
      gemm_strip(hout, w1, H, mb, jc, lane, acc);
      for (int m = 0; m < 2; ++m)
        for (int i = 0; i < 8; ++i) {
          int row = mb + (m << 4) + i + ((lane >> 4) << 3);
          t1h[row * H + jc + ncol] = (_Float16)acc[m][i];
        }
    }
    grid_sync(bar);

    // ===== Phase 3: t2 = t1 @ W2^T + b2 ===================================
    {
      v8f acc[2];
      float bb = b2[jc + ncol];
      for (int m = 0; m < 2; ++m) for (int i = 0; i < 8; ++i) acc[m][i] = bb;
      gemm_strip(t1h, w2, H, mb, jc, lane, acc);
      for (int m = 0; m < 2; ++m)
        for (int i = 0; i < 8; ++i) {
          int row = mb + (m << 4) + i + ((lane >> 4) << 3);
          t2h[row * H + jc + ncol] = (_Float16)acc[m][i];
        }
    }
    grid_sync(bar);

    // ===== Phase 4: out = inp + t2 @ W3^T + b3; prepare next inp ==========
    if ((wave & 15) == 0) {          // 8 units, one per (up to) 8 WGPs
      int u4  = wave >> 4;           // 0..7
      int jc4 = (u4 >> 1) << 4;      // col base 0/16/32/48
      int mb4 = (u4 & 1) << 5;       // batch half
      v8f acc[2];
      for (int m = 0; m < 2; ++m) for (int i = 0; i < 8; ++i) acc[m][i] = 0.f;
      gemm_strip(t2h, w3, H, mb4, jc4, lane, acc);
      int col = jc4 + ncol;
      for (int m = 0; m < 2; ++m)
        for (int i = 0; i < 8; ++i) {
          int row = mb4 + (m << 4) + i + ((lane >> 4) << 3);
          if (col < D) {
            float o = acc[m][i] + inp32[(row << 6) + col] + b3[col];
            out[(row * T + t) * D + col] = o;
            if (t + 1 < T) {
              int c = (((t + 1) % per) < gt) ? 1 : 0;
              float nx = c ? x[(row * T + t + 1) * D + col] : o;
              inp32[(row << 6) + col] = nx;
              inp16[(row << 6) + col] = (_Float16)nx;
            }
          }
        }
    }
    grid_sync(bar);
  }
}

// ---------------------------------------------------------------------------
extern "C" void kernel_launch(void* const* d_in, const int* in_sizes, int n_in,
                              void* d_out, int out_size, void* d_ws, size_t ws_size,
                              hipStream_t stream) {
  const float* x   = (const float*)d_in[0];
  const float* Wih = (const float*)d_in[1];
  const float* Whh = (const float*)d_in[2];
  const float* bih = (const float*)d_in[3];
  const float* bhh = (const float*)d_in[4];
  const float* W1f = (const float*)d_in[5];
  const float* b1  = (const float*)d_in[6];
  const float* W2f = (const float*)d_in[7];
  const float* b2  = (const float*)d_in[8];
  const float* W3f = (const float*)d_in[9];
  const float* b3  = (const float*)d_in[10];
  const int*   cl  = (const int*)d_in[11];  // condition_length
  const int*   gt  = (const int*)d_in[12];  // ground_truth_length

  char* ws = (char*)d_ws;
  size_t off = 0;
  auto take = [&](size_t bytes) -> void* {
    void* p = ws + off;
    off = (off + bytes + 255) & ~(size_t)255;
    return p;
  };
  _Float16* wih16 = (_Float16*)take((size_t)3 * H * DPAD * 2);
  _Float16* whh16 = (_Float16*)take((size_t)3 * H * H * 2);
  _Float16* w1h   = (_Float16*)take((size_t)H * H * 2);
  _Float16* w2h   = (_Float16*)take((size_t)H * H * 2);
  _Float16* w3h   = (_Float16*)take((size_t)DPAD * H * 2);
  float*    h32   = (float*)   take((size_t)B * H * 4);
  _Float16* h16a  = (_Float16*)take((size_t)B * H * 2);
  _Float16* h16b  = (_Float16*)take((size_t)B * H * 2);
  _Float16* t1h   = (_Float16*)take((size_t)B * H * 2);
  _Float16* t2h   = (_Float16*)take((size_t)B * H * 2);
  float*    inp32 = (float*)   take((size_t)B * DPAD * 4);
  _Float16* inp16 = (_Float16*)take((size_t)B * DPAD * 2);
  unsigned* bar   = (unsigned*)take(256);

  prep_weights<<<2048, 256, 0, stream>>>(Wih, Whh, W1f, W2f, W3f,
                                         wih16, whh16, w1h, w2h, w3h);
  init_state<<<256, 256, 0, stream>>>(x, gt, cl, h32, h16a, h16b,
                                      inp32, inp16, bar);
  gru_rollout<<<NWG, THREADS, 0, stream>>>(x, wih16, whh16, bih, bhh,
                                           w1h, b1, w2h, b2, w3h, b3,
                                           (float*)d_out,
                                           h32, h16a, h16b, t1h, t2h,
                                           inp32, inp16, bar, gt, cl);
}
